// MOELayer_45475113730574
// MI455X (gfx1250) — matrix operand order repaired and checked by reading, not compiled, not run on hardware
//
#include <hip/hip_runtime.h>
#include <hip/hip_bf16.h>
#include <math.h>
#include <stdint.h>

typedef __attribute__((ext_vector_type(16))) _Float16 v16h;
typedef __attribute__((ext_vector_type(8)))  _Float16 v8h;
typedef __attribute__((ext_vector_type(8)))  float    v8f;

// Problem dims (from reference setup_inputs)
static constexpr int S_TOK = 4096;           // tokens (1*8*512)
static constexpr int MODEL = 1024;           // m
static constexpr int NEXP  = 16;             // experts
static constexpr int DFF   = 4096;           // ffn hidden
static constexpr int CAP   = 2 * S_TOK / NEXP;   // 512
static constexpr int ROWS  = NEXP * CAP;     // 8192 expert-buffer rows

// ---------------------------------------------------------------------------
// Gating: logits = x @ wg   (4096 x 16). One thread per (token, expert).
// ---------------------------------------------------------------------------
__global__ __launch_bounds__(256) void gate_logits_kernel(
    const float* __restrict__ x, const float* __restrict__ wg,
    float* __restrict__ logits)
{
    int t = blockIdx.x * 16 + (threadIdx.x >> 4);
    int e = threadIdx.x & 15;
    const float* xr = x + (size_t)t * MODEL;
    float acc = 0.f;
    for (int k = 0; k < MODEL; ++k)
        acc = fmaf(xr[k], wg[k * NEXP + e], acc);
    logits[t * NEXP + e] = acc;
}

// ---------------------------------------------------------------------------
// Top-2 selection + softmax gate values (pre-capacity). One thread / token.
// ---------------------------------------------------------------------------
__global__ __launch_bounds__(256) void top2_kernel(
    const float* __restrict__ logits,
    int* __restrict__ idx1, int* __restrict__ idx2,
    float* __restrict__ g1r, float* __restrict__ g2r)
{
    int t = blockIdx.x * blockDim.x + threadIdx.x;
    if (t >= S_TOK) return;
    const float* l = logits + t * NEXP;
    float mx = l[0]; int i1 = 0;
    for (int e = 1; e < NEXP; ++e) if (l[e] > mx) { mx = l[e]; i1 = e; }
    float sum = 0.f;
    for (int e = 0; e < NEXP; ++e) sum += expf(l[e] - mx);
    float m2 = -INFINITY; int i2 = 0;
    for (int e = 0; e < NEXP; ++e)
        if (e != i1 && l[e] > m2) { m2 = l[e]; i2 = e; }
    idx1[t] = i1; idx2[t] = i2;
    g1r[t] = 1.0f / sum;                 // exp(mx-mx)/sum
    g2r[t] = expf(m2 - mx) / sum;
}

// ---------------------------------------------------------------------------
// Sequential per-expert cumsum -> unique slot assignment with capacity drop.
// Single workgroup; thread e owns expert e. slot = e*CAP + pos, or -1.
// ---------------------------------------------------------------------------
__global__ void scan_assign_kernel(
    const int* __restrict__ idx1, const int* __restrict__ idx2,
    int* __restrict__ slot1, int* __restrict__ slot2)
{
    int e = threadIdx.x;
    if (e >= NEXP) return;
    int c1 = 0;
    for (int t = 0; t < S_TOK; ++t)
        if (idx1[t] == e) { int p = c1++; slot1[t] = (p < CAP) ? e * CAP + p : -1; }
    int c2 = 0;
    for (int t = 0; t < S_TOK; ++t)
        if (idx2[t] == e) { int p = (c2++) + c1; slot2[t] = (p < CAP) ? e * CAP + p : -1; }
}

// ---------------------------------------------------------------------------
// Renormalize gates after capacity drops: g = g_kept / max(g1+g2, eps)
// ---------------------------------------------------------------------------
__global__ __launch_bounds__(256) void renorm_kernel(
    const float* __restrict__ g1r, const float* __restrict__ g2r,
    const int* __restrict__ slot1, const int* __restrict__ slot2,
    float* __restrict__ g1, float* __restrict__ g2)
{
    int t = blockIdx.x * blockDim.x + threadIdx.x;
    if (t >= S_TOK) return;
    float a = (slot1[t] >= 0) ? g1r[t] : 0.f;
    float b = (slot2[t] >= 0) ? g2r[t] : 0.f;
    float d = fmaxf(a + b, 1e-9f);
    g1[t] = a / d;
    g2[t] = b / d;
}

// ---------------------------------------------------------------------------
// Dispatch: gather token rows (fp32 -> f16) into expert buffers (row-major).
// ---------------------------------------------------------------------------
__global__ __launch_bounds__(256) void gather_kernel(
    const float* __restrict__ x,
    const int* __restrict__ slot1, const int* __restrict__ slot2,
    _Float16* __restrict__ Xd)
{
    int t = blockIdx.x;
    int s1 = slot1[t], s2 = slot2[t];
    const float* xr = x + (size_t)t * MODEL;
    for (int c = threadIdx.x; c < MODEL; c += 256) {
        _Float16 v = (_Float16)xr[c];
        if (s1 >= 0) Xd[(size_t)s1 * MODEL + c] = v;
        if (s2 >= 0) Xd[(size_t)s2 * MODEL + c] = v;
    }
}

// ---------------------------------------------------------------------------
// Pack a K x N fp32 weight into f16 WMMA B-fragments.
// Tile = 32(K) x 16(N). Fragment layout used here (contiguous k; A loads use the same order):
//   lane L holds column n = L&15; K range (L>>4)*16 .. +15 within the tile.
// Stored contiguously: Bp[((nt*KT + kt)*32 + lane)*16 + e]  (1KB per tile)
// ---------------------------------------------------------------------------
__global__ void pack_b_kernel(
    const float* __restrict__ W, _Float16* __restrict__ Bp, int K, int N)
{
    int tile = blockIdx.x;          // tile = nt*KT + kt
    int KT = K >> 5;
    int nt = tile / KT;
    int kt = tile - nt * KT;
    int lane = threadIdx.x;         // 0..31
    int n  = nt * 16 + (lane & 15);
    int k0 = kt * 32 + ((lane >> 4) << 4);
    _Float16* dst = Bp + (((size_t)tile * 32) + lane) * 16;
    #pragma unroll
    for (int e = 0; e < 16; ++e)
        dst[e] = (_Float16)W[(size_t)(k0 + e) * N + n];
}

// ---------------------------------------------------------------------------
// WMMA GEMM: C(MxN) = A(MxK, f16 row-major) x B(packed f16) + bias
// Block = 8 waves computes a 128x64 tile; wave w owns M-tile w, all waves
// share the same 4 N-tiles. B fragments (4KB per k-step) are double-buffered
// in LDS via async global->LDS copies (ASYNCcnt). All four B fragments are
// preloaded into distinct registers so the 8 ds_load_b128 issue as one group
// and the 4 independent WMMAs drain under a single DScnt wait.
// ---------------------------------------------------------------------------
template <bool RELU_F16_OUT>
__global__ __launch_bounds__(256) void moe_gemm_kernel(
    const _Float16* __restrict__ A, const _Float16* __restrict__ Bp,
    const float* __restrict__ bias,
    _Float16* __restrict__ outH, float* __restrict__ outF,
    int M, int N, int K)
{
    __shared__ __align__(32) _Float16 Bs[2 * 4 * 512];   // 2 x 4KB double buffer

    const int lane = threadIdx.x & 31;
    const int wave = threadIdx.x >> 5;
    const int KT   = K >> 5;
    const int nBN  = N >> 6;                  // blocks along N (64 cols each)
    const int bm   = blockIdx.x / nBN;
    const int bn   = blockIdx.x - bm * nBN;
    const int tm   = bm * 8 + wave;           // this wave's 16-row M tile
    const int nt0  = bn * 4;                  // first 16-col N tile of block

    // Per-thread async-staging coordinates: thread -> (N-tile js, 16B chunk qs)
    const int js = threadIdx.x >> 6;          // 0..3
    const int qs = threadIdx.x & 63;          // 0..63 (16B chunks of 1KB tile)
    const _Float16* bsrc0 = Bp + (size_t)(nt0 + js) * KT * 512 + qs * 8;
    // LDS byte offset of this thread's staging slot (low 32 bits of the
    // LDS-aperture flat address == wave-relative LDS offset per ISA 10.2)
    const unsigned ldst0 = (unsigned)(uintptr_t)(&Bs[0]) + js * 1024 + qs * 16;

    const int m  = lane & 15;
    const int kb = (lane >> 4) << 4;
    const _Float16* arow = A + (size_t)(tm * 16 + m) * K + kb;

    // Stage k-step 0 into buffer 0
    asm volatile("global_load_async_to_lds_b128 %0, %1, off"
                 :: "v"(ldst0), "v"(bsrc0) : "memory");
    asm volatile("s_wait_asynccnt 0x0" ::: "memory");
    __syncthreads();

    v8f acc[4] = {};
    int buf = 0;
    for (int kt = 0; kt < KT; ++kt) {
        // Prefetch next k-step's B fragments into the other LDS buffer
        if (kt + 1 < KT) {
            const _Float16* g = bsrc0 + (size_t)(kt + 1) * 512;
            unsigned l = ldst0 + ((buf ^ 1) << 12);
            asm volatile("global_load_async_to_lds_b128 %0, %1, off"
                         :: "v"(l), "v"(g) : "memory");
        }
        // A fragment from global (per-wave rows)
        union { v16h v; v8h h[2]; } a;
        a.h[0] = *(const v8h*)(arow + (size_t)kt * 32);
        a.h[1] = *(const v8h*)(arow + (size_t)kt * 32 + 8);
        // Preload all four B fragments (distinct registers -> one DS clause)
        const _Float16* bbuf = &Bs[buf * 2048] + lane * 16;
        const v16h b0 = *(const v16h*)(bbuf);
        const v16h b1 = *(const v16h*)(bbuf + 512);
        const v16h b2 = *(const v16h*)(bbuf + 1024);
        const v16h b3 = *(const v16h*)(bbuf + 1536);
        acc[0] = __builtin_amdgcn_wmma_f32_16x16x32_f16(
            false, a.v, false, b0, (short)0, acc[0], false, false);
        acc[1] = __builtin_amdgcn_wmma_f32_16x16x32_f16(
            false, a.v, false, b1, (short)0, acc[1], false, false);
        acc[2] = __builtin_amdgcn_wmma_f32_16x16x32_f16(
            false, a.v, false, b2, (short)0, acc[2], false, false);
        acc[3] = __builtin_amdgcn_wmma_f32_16x16x32_f16(
            false, a.v, false, b3, (short)0, acc[3], false, false);
        // Drain async copy, then block-wide barrier before buffer flip
        asm volatile("s_wait_asynccnt 0x0" ::: "memory");
        __syncthreads();
        buf ^= 1;
    }

    // Epilogue. C/D layout: VGPR r -> row (lane>=16 ? 8 : 0)+r, col lane&15.
    const int rbase = (lane >> 4) << 3;
    #pragma unroll
    for (int j = 0; j < 4; ++j) {
        int n = (nt0 + j) * 16 + (lane & 15);
        float bv = bias[n];
        #pragma unroll
        for (int r = 0; r < 8; ++r) {
            int row = tm * 16 + rbase + r;
            float v = acc[j][r] + bv;
            if (RELU_F16_OUT) {
                v = fmaxf(v, 0.f);
                outH[(size_t)row * N + n] = (_Float16)v;
            } else {
                outF[(size_t)row * N + n] = v;
            }
        }
    }
}

// ---------------------------------------------------------------------------
// Combine: out[t] = g1*Z[slot1] + g2*Z[slot2]  (0 for dropped tokens)
// ---------------------------------------------------------------------------
__global__ __launch_bounds__(256) void combine_kernel(
    const float* __restrict__ Z,
    const float* __restrict__ g1, const float* __restrict__ g2,
    const int* __restrict__ slot1, const int* __restrict__ slot2,
    float* __restrict__ out)
{
    int t = blockIdx.x;
    int s1 = slot1[t], s2 = slot2[t];
    float a = g1[t], b = g2[t];
    for (int c = threadIdx.x; c < MODEL; c += 256) {
        float v = 0.f;
        if (s1 >= 0) v += a * Z[(size_t)s1 * MODEL + c];
        if (s2 >= 0) v += b * Z[(size_t)s2 * MODEL + c];
        out[(size_t)t * MODEL + c] = v;
    }
}

// ---------------------------------------------------------------------------
extern "C" void kernel_launch(void* const* d_in, const int* in_sizes, int n_in,
                              void* d_out, int out_size, void* d_ws, size_t ws_size,
                              hipStream_t stream)
{
    const float* x  = (const float*)d_in[0];   // (4096, 1024)
    const float* wg = (const float*)d_in[1];   // (1024, 16)
    const float* w1 = (const float*)d_in[2];   // (1024, 4096)
    const float* b1 = (const float*)d_in[3];   // (4096,)
    const float* w2 = (const float*)d_in[4];   // (4096, 1024)
    const float* b2 = (const float*)d_in[5];   // (1024,)
    float* out = (float*)d_out;

    // Workspace carve-up (256B aligned slices)
    char* p = (char*)d_ws;
    auto carve = [&](size_t bytes) -> void* {
        void* r = (void*)p;
        p += (bytes + 255) & ~(size_t)255;
        return r;
    };
    float*     logits = (float*)    carve((size_t)S_TOK * NEXP * 4);
    int*       idx1   = (int*)      carve((size_t)S_TOK * 4);
    int*       idx2   = (int*)      carve((size_t)S_TOK * 4);
    float*     g1r    = (float*)    carve((size_t)S_TOK * 4);
    float*     g2r    = (float*)    carve((size_t)S_TOK * 4);
    float*     g1     = (float*)    carve((size_t)S_TOK * 4);
    float*     g2     = (float*)    carve((size_t)S_TOK * 4);
    int*       slot1  = (int*)      carve((size_t)S_TOK * 4);
    int*       slot2  = (int*)      carve((size_t)S_TOK * 4);
    _Float16*  Xd     = (_Float16*) carve((size_t)ROWS * MODEL * 2);   // 16 MB
    _Float16*  Bp1    = (_Float16*) carve((size_t)MODEL * DFF   * 2);  // 8 MB
    _Float16*  Bp2    = (_Float16*) carve((size_t)DFF   * MODEL * 2);  // 8 MB
    _Float16*  H      = (_Float16*) carve((size_t)ROWS * DFF    * 2);  // 64 MB
    float*     Z      = (float*)    carve((size_t)ROWS * MODEL  * 4);  // 32 MB

    // --- weight packing into WMMA fragments (per call; deterministic) ---
    pack_b_kernel<<<(DFF / 16) * (MODEL / 32), 32, 0, stream>>>(w1, Bp1, MODEL, DFF);
    pack_b_kernel<<<(MODEL / 16) * (DFF / 32), 32, 0, stream>>>(w2, Bp2, DFF, MODEL);

    // --- gating ---
    gate_logits_kernel<<<S_TOK / 16, 256, 0, stream>>>(x, wg, logits);
    top2_kernel<<<S_TOK / 256, 256, 0, stream>>>(logits, idx1, idx2, g1r, g2r);
    scan_assign_kernel<<<1, 32, 0, stream>>>(idx1, idx2, slot1, slot2);
    renorm_kernel<<<S_TOK / 256, 256, 0, stream>>>(g1r, g2r, slot1, slot2, g1, g2);

    // --- dispatch ---
    hipMemsetAsync(Xd, 0, (size_t)ROWS * MODEL * 2, stream);
    gather_kernel<<<S_TOK, 256, 0, stream>>>(x, slot1, slot2, Xd);

    // --- expert FFN: H = relu(Xd @ w1 + b1); Z = H @ w2 + b2 ---
    moe_gemm_kernel<true><<<(ROWS / 128) * (DFF / 64), 256, 0, stream>>>(
        Xd, Bp1, b1, H, nullptr, ROWS, DFF, MODEL);
    moe_gemm_kernel<false><<<(ROWS / 128) * (MODEL / 64), 256, 0, stream>>>(
        H, Bp2, b2, nullptr, Z, ROWS, MODEL, DFF);

    // --- combine back to token order ---
    combine_kernel<<<S_TOK, 256, 0, stream>>>(Z, g1, g2, slot1, slot2, out);
}
